// MediumRangeEdge_67302137528722
// MI455X (gfx1250) — matrix operand order repaired and checked by reading, not compiled, not run on hardware
//
#include <hip/hip_runtime.h>

#define DIMC 384
#define Hh 56
#define Ww 56
#define KSEL 10
#define NN (Hh * Ww)          // 3136
#define BB 8
#define SUBW 28
#define PS (SUBW * SUBW)      // 784 points per sublattice
#define NTILES 49             // 784 / 16
#define DROW 800              // padded Ds row stride (floats)
#define INF_ 100000.0f

typedef __attribute__((ext_vector_type(16))) __bf16 v16bf;
typedef __attribute__((ext_vector_type(8)))  float  v8f;

__device__ __forceinline__ unsigned short f2bf(float f) {
    unsigned u = __float_as_uint(f);
    unsigned r = u + 0x7FFFu + ((u >> 16) & 1u);   // round-to-nearest-even
    return (unsigned short)(r >> 16);
}

// ---------------------------------------------------------------------------
// Closed-form relative-position table: f[d] = sum_k cos(2*d * 10000^{-k/96})
// rp(dh,dw) = -(f[|dh|] + f[|dw|]) / 192
// ---------------------------------------------------------------------------
__global__ void ftab_kernel(float* __restrict__ ftab) {
    int d = threadIdx.x;
    if (d < SUBW) {
        float delta = 2.0f * (float)d;
        float acc = 0.0f;
        for (int k = 0; k < 96; ++k) {
            float omega = expf(-((float)k / 96.0f) * logf(10000.0f));
            acc += cosf(delta * omega);
        }
        ftab[d] = acc;
    }
}

// ---------------------------------------------------------------------------
// Normalize rows, convert to bf16, scatter into per-(batch,sublattice) panels.
// One wave (32 lanes) per row of 384 floats.
// ---------------------------------------------------------------------------
__global__ __launch_bounds__(256)
void norm_pack_kernel(const float* __restrict__ in,
                      unsigned short* __restrict__ apack,
                      float* __restrict__ sq) {
    int wave = threadIdx.x >> 5;
    int lane = threadIdx.x & 31;
    int row  = blockIdx.x * 8 + wave;
    if (row >= BB * NN) return;

    const float* src = in + (long long)row * DIMC;
    float x[12];
    float ss = 0.0f;
#pragma unroll
    for (int t = 0; t < 12; ++t) {
        x[t] = src[lane + 32 * t];
        ss += x[t] * x[t];
    }
#pragma unroll
    for (int off = 16; off; off >>= 1) ss += __shfl_xor(ss, off, 32);

    float scale = 1.0f / fmaxf(sqrtf(ss), 1e-12f);

    int b = row / NN, id = row % NN;
    int r = id / Ww, c = id % Ww;
    int s = ((r & 1) << 1) | (c & 1);            // sublattice id
    int p = (r >> 1) * SUBW + (c >> 1);          // position within sublattice
    long long base = ((long long)(b * 4 + s) * PS + p) * DIMC;

#pragma unroll
    for (int t = 0; t < 12; ++t)
        apack[base + lane + 32 * t] = f2bf(x[t] * scale);
    if (lane == 0)
        sq[(b * 4 + s) * PS + p] = ss * scale * scale;   // ~1.0, kept exact
}

// ---------------------------------------------------------------------------
// Per (batch, sublattice, 16-row tile): bf16 WMMA GEMM panel + fused distance
// epilogue + iterative top-10 selection, write int32 edge pairs.
// ---------------------------------------------------------------------------
__global__ __launch_bounds__(256)
void dist_topk_kernel(const unsigned short* __restrict__ apack,
                      const float* __restrict__ sq,
                      const float* __restrict__ ftab,
                      int* __restrict__ out) {
    __shared__ unsigned short As[16 * DIMC];     // 12 KB   A row-panel (bf16)
    __shared__ float Ds[16 * DROW];              // 51.2 KB distance row-panel
    __shared__ float Fs[SUBW];
    __shared__ float SqR[16];

    int bs   = blockIdx.x / NTILES;              // batch*4 + sublattice
    int tile = blockIdx.x % NTILES;
    int b = bs >> 2, s = bs & 3;
    int rowBase = tile * 16;

    int tid  = threadIdx.x;
    int wave = tid >> 5, lane = tid & 31;
    int half = lane >> 4, l16 = lane & 15;

    const unsigned short* panel = apack + (long long)bs * PS * DIMC;
    const float* sqp = sq + bs * PS;

    // stage A row-panel into LDS (32-bit granularity: 3072 dwords)
    {
        const unsigned* asrc = (const unsigned*)(panel + (long long)rowBase * DIMC);
        unsigned* adst = (unsigned*)As;
        for (int i = tid; i < 16 * DIMC / 2; i += 256) adst[i] = asrc[i];
    }
    if (tid < SUBW) Fs[tid] = ftab[tid];
    if (tid < 16)   SqR[tid] = sqp[rowBase + tid];
    __syncthreads();

    // -------- GEMM + epilogue: each wave owns column tiles round-robin ------
    for (int ct = wave; ct < NTILES; ct += 8) {
        int colBase = ct * 16;
        const unsigned short* Bb = panel + (long long)colBase * DIMC;

        v8f C = {0.f, 0.f, 0.f, 0.f, 0.f, 0.f, 0.f, 0.f};
#pragma unroll
        for (int kk = 0; kk < DIMC; kk += 32) {
            union { unsigned u[8]; v16bf v; } af, bf;
            // A 16x32 bf16 layout (ISA 7.12.2): lanes0-15 K{0..7,16..23}, lanes16-31 K{8..15,24..31}
#pragma unroll
            for (int v = 0; v < 4; ++v) {
                af.u[v]     = *(const unsigned*)&As[l16 * DIMC + kk + half * 8 + 2 * v];
                af.u[4 + v] = *(const unsigned*)&As[l16 * DIMC + kk + 16 + half * 8 + 2 * v];
            }
            // B 32x16 bf16 layout: lanes0-15 K{0..7,16..23}, lanes16-31 K{8..15,24..31}, col = lane%16
#pragma unroll
            for (int v = 0; v < 4; ++v) {
                bf.u[v]     = *(const unsigned*)&Bb[l16 * DIMC + kk + half * 8 + 2 * v];
                bf.u[4 + v] = *(const unsigned*)&Bb[l16 * DIMC + kk + 16 + half * 8 + 2 * v];
            }

            C = __builtin_amdgcn_wmma_f32_16x16x32_bf16(
                    false, af.v, false, bf.v, (short)0, C, false, false);
        }

        // fused distance epilogue:  sq_i + sq_j - 2*dot + rp(|dh|,|dw|)  (+INF self)
        int pj  = colBase + l16;
        int rj2 = pj / SUBW, cj2 = pj % SUBW;
        float sqj = sqp[pj];
#pragma unroll
        for (int v = 0; v < 8; ++v) {
            int m  = v + 8 * half;
            int pi = rowBase + m;
            int ri2 = pi / SUBW, ci2 = pi % SUBW;
            int dh = (ri2 > rj2) ? (ri2 - rj2) : (rj2 - ri2);
            int dw = (ci2 > cj2) ? (ci2 - cj2) : (cj2 - ci2);
            float rp = -(Fs[dh] + Fs[dw]) * (1.0f / 192.0f);
            float d  = SqR[m] + sqj - 2.0f * C[v] + rp;
            if (pi == pj) d += INF_;                    // self-loop mask
            Ds[m * DROW + pj] = d;
        }
    }
    __syncthreads();

    // -------- top-10 per row: threshold-based iterative selection -----------
    int oh = s >> 1, ow = s & 1;
    for (int rr = 0; rr < 2; ++rr) {
        int m  = wave * 2 + rr;                          // 8 waves x 2 rows = 16
        int pi = rowBase + m;
        int node_i = ((pi / SUBW) * 2 + oh) * Ww + (pi % SUBW) * 2 + ow;
        long long obase = (long long)(b * NN + node_i) * (KSEL * 2);

        float lastv = -3.4e38f;
        int   lasti = -1;
        for (int t = 0; t < KSEL; ++t) {
            float bv = 3.4e38f;
            int   bi = 0x7fffffff;
#pragma unroll
            for (int slot = 0; slot < 25; ++slot) {
                int j = lane + 32 * slot;
                if (j < PS) {
                    float v = Ds[m * DROW + j];
                    // strictly after last pick in (val, idx) lexicographic order
                    bool gt = (v > lastv) || (v == lastv && j > lasti);
                    if (gt && (v < bv || (v == bv && j < bi))) { bv = v; bi = j; }
                }
            }
            // wave32 argmin reduction (lowest index on ties, like lax.top_k)
#pragma unroll
            for (int off = 16; off; off >>= 1) {
                float ov = __shfl_xor(bv, off, 32);
                int   oi = __shfl_xor(bi, off, 32);
                if (ov < bv || (ov == bv && oi < bi)) { bv = ov; bi = oi; }
            }
            lastv = bv; lasti = bi;
            if (lane == 0) {
                int node_j = ((bi / SUBW) * 2 + oh) * Ww + (bi % SUBW) * 2 + ow;
                out[obase + t * 2 + 0] = b * NN + node_j;   // neighbor (dst)
                out[obase + t * 2 + 1] = b * NN + node_i;   // src
            }
        }
    }
}

// ---------------------------------------------------------------------------
extern "C" void kernel_launch(void* const* d_in, const int* in_sizes, int n_in,
                              void* d_out, int out_size, void* d_ws, size_t ws_size,
                              hipStream_t stream) {
    const float* node_feature = (const float*)d_in[0];
    char* ws = (char*)d_ws;

    // workspace layout (all offsets 128B-aligned)
    float* ftab = (float*)ws;                                     // 28 floats
    float* sqv  = (float*)(ws + 128);                             // 8*4*784 floats = 100352 B
    unsigned short* apack = (unsigned short*)(ws + 100608);       // 8*4*784*384 bf16 = 19.27 MB

    ftab_kernel<<<1, 32, 0, stream>>>(ftab);
    norm_pack_kernel<<<(BB * NN) / 8, 256, 0, stream>>>(node_feature, apack, sqv);
    dist_topk_kernel<<<BB * 4 * NTILES, 256, 0, stream>>>(apack, sqv, ftab, (int*)d_out);
}
